// CausalMultiheadSelfAttention_43396349559460
// MI455X (gfx1250) — compile-verified, matrix operand order checked by reading, not run on hardware
//
#include <hip/hip_runtime.h>
#include <hip/hip_bf16.h>
#include <stdint.h>

// ---------------------------------------------------------------------------
// Problem constants (match reference)
// ---------------------------------------------------------------------------
#define BATCH   2
#define SEQ     2048
#define DMODEL  1024
#define HEADS   16
#define DK      64
#define ROWS    (BATCH * SEQ)          // 4096 rows for the big GEMMs
#define KDIM    1024
#define LDSTR   40   // LDS row stride in bf16 elems (80B, 16B-aligned rows)

typedef __attribute__((ext_vector_type(16))) __bf16 v16bf;
typedef __attribute__((ext_vector_type(8)))  float  v8f;
typedef int v4i_gen __attribute__((vector_size(16)));   // generic-vector int4 (builtin param type)

// ---------------------------------------------------------------------------
// Feature probes
// ---------------------------------------------------------------------------
#if __has_builtin(__builtin_amdgcn_permlane16)
#define PERMLANE16_FN __builtin_amdgcn_permlane16
#define HAVE_PERMLANE16 1
#elif __has_builtin(__builtin_amdgcn_permlane16_b32)
#define PERMLANE16_FN __builtin_amdgcn_permlane16_b32
#define HAVE_PERMLANE16 1
#else
#warning "permlane16 builtin not available; falling back to __shfl_xor"
#endif

#if __has_builtin(__builtin_amdgcn_global_load_async_to_lds_b128)
#define HAVE_ASYNC_LDS 1
#else
#warning "global_load_async_to_lds_b128 builtin not available; using sync LDS staging"
#endif

// ---------------------------------------------------------------------------
// bf16 helpers (round-to-nearest-even)
// ---------------------------------------------------------------------------
__device__ __forceinline__ unsigned short f2bf(float f) {
    unsigned u = __float_as_uint(f);
    u += 0x7FFFu + ((u >> 16) & 1u);
    return (unsigned short)(u >> 16);
}
__device__ __forceinline__ float bf2f(unsigned short h) {
    return __uint_as_float(((unsigned)h) << 16);
}

// ---------------------------------------------------------------------------
// 16-lane butterfly reductions (within each half-wave) via v_permlane16_b32.
// Select tables: nibble i of {hi,lo} = i ^ mask, identical for both rows.
// ---------------------------------------------------------------------------
#ifdef HAVE_PERMLANE16
__device__ __forceinline__ float bfly16(float v, unsigned lo, unsigned hi) {
    unsigned u = __float_as_uint(v);
    return __uint_as_float(PERMLANE16_FN(u, u, lo, hi, false, false));
}
__device__ __forceinline__ float red_max16(float x) {
    x = fmaxf(x, bfly16(x, 0x67452301u, 0xEFCDAB89u));   // xor 1
    x = fmaxf(x, bfly16(x, 0x54761032u, 0xDCFE98BAu));   // xor 2
    x = fmaxf(x, bfly16(x, 0x32107654u, 0xBA98FEDCu));   // xor 4
    x = fmaxf(x, bfly16(x, 0xFEDCBA98u, 0x76543210u));   // xor 8
    return x;
}
__device__ __forceinline__ float red_add16(float x) {
    x += bfly16(x, 0x67452301u, 0xEFCDAB89u);
    x += bfly16(x, 0x54761032u, 0xDCFE98BAu);
    x += bfly16(x, 0x32107654u, 0xBA98FEDCu);
    x += bfly16(x, 0xFEDCBA98u, 0x76543210u);
    return x;
}
#else
__device__ __forceinline__ float red_max16(float x) {
    for (int off = 1; off < 16; off <<= 1) x = fmaxf(x, __shfl_xor(x, off, 16));
    return x;
}
__device__ __forceinline__ float red_add16(float x) {
    for (int off = 1; off < 16; off <<= 1) x += __shfl_xor(x, off, 16);
    return x;
}
#endif

// ---------------------------------------------------------------------------
// Async global -> LDS staging (ASYNCcnt path) with sync fallback.
// Builtin prototype (from clang diagnostic): (v4i AS1*, v4i AS3*, Ii, Ii).
// ---------------------------------------------------------------------------
__device__ __forceinline__ void stage_b128(const unsigned short* g, unsigned short* l) {
#ifdef HAVE_ASYNC_LDS
    __builtin_amdgcn_global_load_async_to_lds_b128(
        (__attribute__((address_space(1))) v4i_gen*)g,
        (__attribute__((address_space(3))) v4i_gen*)l,
        0, 0);
#else
    *(uint4*)l = *(const uint4*)g;
#endif
}

template <int N>
__device__ __forceinline__ void stage_wait() {
#ifdef HAVE_ASYNC_LDS
    asm volatile("s_wait_asynccnt %0" :: "i"(N) : "memory");
#endif
}

// ---------------------------------------------------------------------------
// WMMA fragment loader.
// CDNA5 16-bit A layout (16x32, MxK): lane l holds row (l&15); with g=l>>4,
// VGPR0..3 hold K = 8g..8g+7 and VGPR4..7 hold K = 16+8g..+7. The B operand
// (32x16, KxN) mirrors this (lane holds column l&15, same K packing), so any
// row-major [N,K] NT operand loads with the identical pattern: two 16-byte
// loads per lane.
// ---------------------------------------------------------------------------
__device__ __forceinline__ v16bf frag_ld(const unsigned short* base, int stride) {
    int lane = threadIdx.x & 31;
    int r = lane & 15, g = lane >> 4;
    const unsigned short* p = base + (size_t)r * stride + 8 * g;
    union { struct { uint4 lo, hi; } fp; v16bf v; } u;
    u.fp.lo = *(const uint4*)(p);
    u.fp.hi = *(const uint4*)(p + 16);
    return u.v;
}

__device__ __forceinline__ v8f wmma_bf16(v16bf a, v16bf b, v8f c) {
    return __builtin_amdgcn_wmma_f32_16x16x32_bf16(
        /*neg_a=*/false, a, /*neg_b=*/false, b,
        /*c_mod=*/(short)0, c, /*reuse_a=*/false, /*reuse_b=*/false);
}

// ---------------------------------------------------------------------------
// f32 -> bf16 bulk conversion (one-shot; removes cvt work from GEMM K-loops)
// ---------------------------------------------------------------------------
__global__ __launch_bounds__(256) void cvt_bf16_kernel(const float* __restrict__ src,
                                                       unsigned short* __restrict__ dst,
                                                       int n4) {
    int i = blockIdx.x * 256 + threadIdx.x;
    if (i >= n4) return;
    float4 f = ((const float4*)src)[i];
    uint2 pk;
    pk.x = (unsigned)f2bf(f.x) | ((unsigned)f2bf(f.y) << 16);
    pk.y = (unsigned)f2bf(f.z) | ((unsigned)f2bf(f.w) << 16);
    ((uint2*)dst)[i] = pk;
}

// ---------------------------------------------------------------------------
// NT GEMM (all-bf16 operands): out[i,j] = sum_k A[i,k] * W[j,k], K = 1024.
// Block: 256 threads (8 waves), 128x128 tile, K-step 32, double-buffered
// async LDS staging. Wave (wm=w>>1, wn=w&1) computes rows wm*32..+31,
// cols wn*64..+63 => 8 WMMAs per K-step per wave.
// ---------------------------------------------------------------------------
template <bool OUT_BF16>
__global__ __launch_bounds__(256) void gemm_nt_kernel(const unsigned short* __restrict__ A,
                                                      const unsigned short* __restrict__ W,
                                                      void* __restrict__ outp) {
    __shared__ unsigned short As[2][128 * LDSTR];
    __shared__ unsigned short Bs[2][128 * LDSTR];

    const int tid  = threadIdx.x;
    const int wv   = tid >> 5;
    const int lane = tid & 31;
    const int r16  = lane & 15;
    const int g    = lane >> 4;
    const int wm   = wv >> 1;      // 0..3
    const int wn   = wv & 1;       // 0..1

    const size_t arow0 = (size_t)blockIdx.x * 128;
    const size_t brow0 = (size_t)blockIdx.y * 128;

    // Per-thread staging slots: idx = tid + 256*i, rr = idx>>2, c8 = (idx&3)*8
    auto issue = [&](int buf, int kk0) {
#pragma unroll
        for (int i = 0; i < 2; ++i) {
            int idx = tid + i * 256;            // 0..511
            int rr  = idx >> 2;                 // 0..127
            int c8  = (idx & 3) * 8;            // 0,8,16,24
            stage_b128(A + (arow0 + rr) * KDIM + kk0 + c8, &As[buf][rr * LDSTR + c8]);
            stage_b128(W + (brow0 + rr) * KDIM + kk0 + c8, &Bs[buf][rr * LDSTR + c8]);
        }
    };

    v8f acc[2][4];
#pragma unroll
    for (int i = 0; i < 2; ++i)
#pragma unroll
        for (int j = 0; j < 4; ++j) acc[i][j] = 0.0f;

    issue(0, 0);

    const int NKT = KDIM / 32;                  // 32 K-steps
    for (int kt = 0; kt < NKT; ++kt) {
        const int buf = kt & 1;
        if (kt + 1 < NKT) {
            issue(buf ^ 1, (kt + 1) * 32);
            stage_wait<4>();                    // tile kt's 4 async ops done
        } else {
            stage_wait<0>();
        }
        __syncthreads();

        v16bf af[2], bfm[4];
#pragma unroll
        for (int i = 0; i < 2; ++i)
            af[i] = frag_ld(&As[buf][(wm * 32 + 16 * i) * LDSTR], LDSTR);
#pragma unroll
        for (int j = 0; j < 4; ++j)
            bfm[j] = frag_ld(&Bs[buf][(wn * 64 + 16 * j) * LDSTR], LDSTR);
#pragma unroll
        for (int i = 0; i < 2; ++i)
#pragma unroll
            for (int j = 0; j < 4; ++j)
                acc[i][j] = wmma_bf16(af[i], bfm[j], acc[i][j]);
        __syncthreads();                        // protect buf^1 (WAR) before next issue
    }

    // ---- epilogue: C/D layout is row m = rr + 8g, col n = r16 ----
#pragma unroll
    for (int i = 0; i < 2; ++i)
#pragma unroll
        for (int j = 0; j < 4; ++j)
#pragma unroll
            for (int rr = 0; rr < 8; ++rr) {
                size_t m = arow0 + wm * 32 + 16 * i + rr + 8 * g;
                size_t n = brow0 + wn * 64 + 16 * j + r16;
                float v  = acc[i][j][rr];
                if constexpr (OUT_BF16)
                    ((unsigned short*)outp)[m * KDIM + n] = f2bf(v);
                else
                    ((float*)outp)[m * KDIM + n] = v;
            }
}

// ---------------------------------------------------------------------------
// RoPE on Q,K (in place, bf16) + V transpose to vT[b, h, dk, s].
// ---------------------------------------------------------------------------
__global__ __launch_bounds__(256) void rope_vt_kernel(unsigned short* __restrict__ q,
                                                      unsigned short* __restrict__ k,
                                                      const unsigned short* __restrict__ v,
                                                      unsigned short* __restrict__ vT,
                                                      const int* __restrict__ pos) {
    unsigned tid = blockIdx.x * blockDim.x + threadIdx.x;  // 2^21 total
    int i = tid & 31;
    int h = (tid >> 5) & 15;
    int s = (tid >> 9) & 2047;
    int b = tid >> 20;

    float p   = (float)pos[s];
    // inv_freq = theta^(-2i/dk) = exp(-(2i/64)*ln(10000))
    float inv = __expf(-((float)(2 * i) * (1.0f / 64.0f)) * 9.210340371976184f);
    float c, sn;
    __sincosf(p * inv, &sn, &c);

    size_t base = ((size_t)(b * SEQ + s)) * DMODEL + h * DK + 2 * i;

    float qe = bf2f(q[base]), qo = bf2f(q[base + 1]);
    q[base]     = f2bf(qe * c - qo * sn);
    q[base + 1] = f2bf(qo * c + qe * sn);

    float ke = bf2f(k[base]), ko = bf2f(k[base + 1]);
    k[base]     = f2bf(ke * c - ko * sn);
    k[base + 1] = f2bf(ko * c + ke * sn);

    size_t vt0 = (((size_t)(b * HEADS + h)) * DK + 2 * i) * SEQ + s;
    vT[vt0]       = v[base];
    vT[vt0 + SEQ] = v[base + 1];
}

// ---------------------------------------------------------------------------
// Flash attention, causal, scale = 1/8.
// Grid: BATCH*HEADS*(SEQ/64) blocks, 128 threads (4 waves), wave owns 16
// query rows. Per 32-wide K tile: 4 score WMMAs, online softmax using
// permlane16 butterflies, P transposed via per-wave LDS, 4 PV WMMAs.
// ---------------------------------------------------------------------------
#define PSTR 48   // per-wave P scratch row stride (96B, 16B-aligned rows)

__global__ __launch_bounds__(128) void flash_attn_kernel(const unsigned short* __restrict__ q,
                                                         const unsigned short* __restrict__ k,
                                                         const unsigned short* __restrict__ vT,
                                                         unsigned short* __restrict__ o) {
    __shared__ unsigned short pshm[4][16 * PSTR];

    const int QT  = SEQ / 64;
    int qt  = blockIdx.x % QT;
    int bh  = blockIdx.x / QT;
    int h   = bh % HEADS;
    int b   = bh / HEADS;

    int wave = threadIdx.x >> 5;
    int lane = threadIdx.x & 31;
    int r16  = lane & 15, g = lane >> 4;

    const int q0 = qt * 64 + wave * 16;   // wave's query rows q0..q0+15

    const unsigned short* qbase = q  + ((size_t)(b * SEQ + q0)) * DMODEL + h * DK;
    const unsigned short* kbase = k  + ((size_t)(b * SEQ)) * DMODEL + h * DK;
    const unsigned short* vtb   = vT + ((size_t)(b * HEADS + h)) * DK * SEQ;

    v16bf qa[2];
    qa[0] = frag_ld(qbase + 0,  DMODEL);
    qa[1] = frag_ld(qbase + 32, DMODEL);

    float mrow[8], lrow[8];
    v8f oacc[4];
#pragma unroll
    for (int rr = 0; rr < 8; ++rr) { mrow[rr] = -__builtin_inff(); lrow[rr] = 0.0f; }
#pragma unroll
    for (int j = 0; j < 4; ++j) oacc[j] = 0.0f;

    unsigned short* ps = pshm[wave];
    const float scale = 0.125f;  // 1/sqrt(64)
    const int ktmax = (qt * 64 + 63) >> 5;

    for (int kt = 0; kt <= ktmax; ++kt) {
        const int kk0 = kt * 32;

        // ---- scores: S[16q x 32k] = Q (16x64) . K^T ----
        v8f s0 = 0.0f, s1 = 0.0f;
#pragma unroll
        for (int dc = 0; dc < 2; ++dc) {
            v16bf kb0 = frag_ld(kbase + (size_t)(kk0)      * DMODEL + dc * 32, DMODEL);
            v16bf kb1 = frag_ld(kbase + (size_t)(kk0 + 16) * DMODEL + dc * 32, DMODEL);
            s0 = wmma_bf16(qa[dc], kb0, s0);
            s1 = wmma_bf16(qa[dc], kb1, s1);
        }

        // ---- online softmax (rows m = rr + 8g; cols kk0+r16, kk0+16+r16) ----
        float p0[8], p1[8];
#pragma unroll
        for (int rr = 0; rr < 8; ++rr) {
            int qg = q0 + 8 * g + rr;
            float a  = s0[rr] * scale;
            float bb = s1[rr] * scale;
            if (kk0 + r16      > qg) a  = -__builtin_inff();
            if (kk0 + 16 + r16 > qg) bb = -__builtin_inff();

            float mx = red_max16(fmaxf(a, bb));

            float mo = mrow[rr];
            float mn = fmaxf(mo, mx);
            float corr = __expf(mo - mn);
            mrow[rr] = mn;

            float e0 = __expf(a - mn);
            float e1 = __expf(bb - mn);
            float rs = red_add16(e0 + e1);

            lrow[rr] = lrow[rr] * corr + rs;
            p0[rr] = e0;
            p1[rr] = e1;
#pragma unroll
            for (int j = 0; j < 4; ++j) oacc[j][rr] *= corr;
        }

        // ---- transpose P through LDS into A-layout ----
#pragma unroll
        for (int rr = 0; rr < 8; ++rr) {
            int m = rr + 8 * g;
            ps[m * PSTR + r16]      = f2bf(p0[rr]);
            ps[m * PSTR + 16 + r16] = f2bf(p1[rr]);
        }
        asm volatile("s_wait_dscnt 0x0" ::: "memory");
        v16bf pa = frag_ld(ps, PSTR);

        // ---- O += P (16x32) . V (32x64), via vT rows (contiguous along s) ----
#pragma unroll
        for (int j = 0; j < 4; ++j) {
            v16bf vb = frag_ld(vtb + (size_t)(j * 16) * SEQ + kk0, SEQ);
            oacc[j] = wmma_bf16(pa, vb, oacc[j]);
        }
        asm volatile("s_wait_dscnt 0x0" ::: "memory");
    }

    // ---- finalize: O /= l, write bf16 [b, s, h*64 + d] ----
    float linv[8];
#pragma unroll
    for (int rr = 0; rr < 8; ++rr) linv[rr] = 1.0f / lrow[rr];

    unsigned short* ob = o + ((size_t)(b * SEQ + q0)) * DMODEL + h * DK;
#pragma unroll
    for (int j = 0; j < 4; ++j)
#pragma unroll
        for (int rr = 0; rr < 8; ++rr) {
            int m = rr + 8 * g;
            ob[(size_t)m * DMODEL + j * 16 + r16] = f2bf(oacc[j][rr] * linv[rr]);
        }
}

// ---------------------------------------------------------------------------
// Launch
// ---------------------------------------------------------------------------
extern "C" void kernel_launch(void* const* d_in, const int* in_sizes, int n_in,
                              void* d_out, int out_size, void* d_ws, size_t ws_size,
                              hipStream_t stream) {
    const float* x  = (const float*)d_in[0];
    const float* Wq = (const float*)d_in[1];
    const float* Wk = (const float*)d_in[2];
    const float* Wv = (const float*)d_in[3];
    const float* Wo = (const float*)d_in[4];
    const int*  pos = (const int*)d_in[5];
    float* out = (float*)d_out;

    // Workspace partition (bf16 elements)
    const size_t NX = (size_t)ROWS * DMODEL;     // 4M elems (x-shaped)
    const size_t NW = (size_t)DMODEL * DMODEL;   // 1M elems (weight-shaped)
    unsigned short* qb  = (unsigned short*)d_ws;
    unsigned short* kb  = qb  + NX;
    unsigned short* vb  = kb  + NX;
    unsigned short* vT  = vb  + NX;
    unsigned short* ab  = vT  + NX;   // attention output (bf16)
    unsigned short* xb  = ab  + NX;   // x in bf16
    unsigned short* wqb = xb  + NX;
    unsigned short* wkb = wqb + NW;
    unsigned short* wvb = wkb + NW;
    unsigned short* wob = wvb + NW;

    // One-shot bf16 conversions
    cvt_bf16_kernel<<<(int)(NX / 4 / 256), 256, 0, stream>>>(x,  xb,  (int)(NX / 4));
    cvt_bf16_kernel<<<(int)(NW / 4 / 256), 256, 0, stream>>>(Wq, wqb, (int)(NW / 4));
    cvt_bf16_kernel<<<(int)(NW / 4 / 256), 256, 0, stream>>>(Wk, wkb, (int)(NW / 4));
    cvt_bf16_kernel<<<(int)(NW / 4 / 256), 256, 0, stream>>>(Wv, wvb, (int)(NW / 4));
    cvt_bf16_kernel<<<(int)(NW / 4 / 256), 256, 0, stream>>>(Wo, wob, (int)(NW / 4));

    dim3 ggrid(ROWS / 128, DMODEL / 128);   // 32 x 8
    gemm_nt_kernel<true><<<ggrid, 256, 0, stream>>>(xb, wqb, qb);
    gemm_nt_kernel<true><<<ggrid, 256, 0, stream>>>(xb, wkb, kb);
    gemm_nt_kernel<true><<<ggrid, 256, 0, stream>>>(xb, wvb, vb);

    int rope_threads = BATCH * SEQ * HEADS * (DK / 2);   // 2^21
    rope_vt_kernel<<<rope_threads / 256, 256, 0, stream>>>(qb, kb, vb, vT, pos);

    int fblocks = BATCH * HEADS * (SEQ / 64);            // 1024
    flash_attn_kernel<<<fblocks, 128, 0, stream>>>(qb, kb, vT, ab);

    gemm_nt_kernel<false><<<ggrid, 256, 0, stream>>>(ab, wob, out);
}
